// SegmentCorrelation3_57810259804529
// MI455X (gfx1250) — matrix operand order repaired and checked by reading, not compiled, not run on hardware
//
#include <hip/hip_runtime.h>

typedef __attribute__((ext_vector_type(16))) _Float16 v16h;
typedef __attribute__((ext_vector_type(8)))  _Float16 v8h;
typedef __attribute__((ext_vector_type(4)))  _Float16 v4h;
typedef __attribute__((ext_vector_type(8)))  float    v8f;
typedef __attribute__((ext_vector_type(4)))  float    v4f;
typedef __attribute__((ext_vector_type(2)))  float    v2f;

union V16 { v16h v; v8h h[2]; v4h h4[4]; };

#define NKEY 255          // M-1 valid key segments
#define SEG_STRIDE 8192   // LS*H*D
#define L_STRIDE 512      // H*D
#define B_STRIDE 2097152  // L*H*D

// LDS layout (byte offsets into one arena; regions overlap across phases)
#define AROW 264   // A (softmax probs) row stride in halfs (256 + 8 pad)
#define QROW 136   // Q chunk row stride in halfs (128 + 8 pad)
#define KROW 136
#define VROW 36    // V-transposed row stride in halfs (32 + 4 pad), 72B, 8B-aligned reads
#define AS_OFF 0        // f16 [16][264]   = 8448 B   (alive: softmax -> end)
#define SS_OFF 8448     // f32 [16][256]   = 16384 B  (alive: pass1 -> softmax)
#define QS_OFF 24832    // f16 [16][136]   = 4352 B   (pass1)
#define KS_OFF 29184    // f16 [64][136]   = 17408 B  (pass1)
#define VS_OFF 8448     // f16 [512][36]   = 36864 B  (pass2, overlaps Ss/Qs/Ks)
#define RED_OFF 46592   // f32 [16][8]     = 512 B    (softmax scratch)
#define SMEM_BYTES 49408

__global__ __launch_bounds__(128)
void seg_corr3_kernel(const float* __restrict__ q, const float* __restrict__ k,
                      const float* __restrict__ v, float* __restrict__ out)
{
    __shared__ __align__(16) char smem[SMEM_BYTES];
    _Float16* As = (_Float16*)(smem + AS_OFF);
    float*    Ss = (float*)   (smem + SS_OFF);
    _Float16* Qs = (_Float16*)(smem + QS_OFF);
    _Float16* Ks = (_Float16*)(smem + KS_OFF);
    _Float16* Vs = (_Float16*)(smem + VS_OFF);
    float*    red = (float*)  (smem + RED_OFF);

    const int tid  = threadIdx.x;
    const int wave = tid >> 5;
    const int lane = tid & 31;
    const int bid  = blockIdx.x;
    const int mt   = bid & 15;          // 16-row tile of M=256
    const int h    = (bid >> 4) & 7;
    const int b    = bid >> 7;

    const size_t base = (size_t)b * B_STRIDE + (size_t)h * 64;

    const int arow  = lane & 15;   // A-frag row / B-frag column / D column
    const int ahalf = lane >> 4;   // lane-half selects K sub-ranges per ISA layout

    // ---------------- Pass 1: scores = Qseg . Kseg^T  (K-dim 1024 in chunks of 128)
    for (int ng = 0; ng < 4; ++ng) {           // 4 groups of 64 key segments
        v8f acc = {};
        for (int kc = 0; kc < 8; ++kc) {
            __syncthreads();
            // stage Q chunk [16 rows][128 kdims]: float4 loads, v4h stores
            #pragma unroll
            for (int i = 0; i < 4; ++i) {
                int f = i * 128 + tid;          // float4 index, 0..511
                int row = f >> 5;               // 0..15
                int kk = (f & 31) * 4;          // 0..124
                int kd = kc * 128 + kk;
                v4f val = *(const v4f*)&q[base + (size_t)(mt * 16 + row) * SEG_STRIDE
                                          + (kd >> 6) * L_STRIDE + (kd & 63)];
                v4h hv;
                hv[0] = (_Float16)val[0]; hv[1] = (_Float16)val[1];
                hv[2] = (_Float16)val[2]; hv[3] = (_Float16)val[3];
                *(v4h*)&Qs[row * QROW + kk] = hv;
            }
            // stage K chunk [64 key rows][128 kdims]
            #pragma unroll
            for (int i = 0; i < 16; ++i) {
                int f = i * 128 + tid;          // float4 index, 0..2047
                int row = f >> 5;               // 0..63
                int kk = (f & 31) * 4;
                int n = ng * 64 + row;
                int seg = n < NKEY ? n : NKEY - 1;   // clamp pad column (masked later)
                int kd = kc * 128 + kk;
                v4f val = *(const v4f*)&k[base + (size_t)seg * SEG_STRIDE
                                          + (kd >> 6) * L_STRIDE + (kd & 63)];
                v4h hv;
                hv[0] = (_Float16)val[0]; hv[1] = (_Float16)val[1];
                hv[2] = (_Float16)val[2]; hv[3] = (_Float16)val[3];
                *(v4h*)&Ks[row * KROW + kk] = hv;
            }
            __syncthreads();
            #pragma unroll
            for (int kw = 0; kw < 4; ++kw) {     // 4 x K=32 per 128-chunk
                int kb = kw * 32;
                V16 a, bm;
                a.h[0] = *(const v8h*)&Qs[arow * QROW + kb + ahalf * 8];
                a.h[1] = *(const v8h*)&Qs[arow * QROW + kb + 16 + ahalf * 8];
                int krow = wave * 16 + arow;
                bm.h[0] = *(const v8h*)&Ks[krow * KROW + kb + ahalf * 8];
                bm.h[1] = *(const v8h*)&Ks[krow * KROW + kb + 16 + ahalf * 8];
                acc = __builtin_amdgcn_wmma_f32_16x16x32_f16(
                          false, a.v, false, bm.v, (short)0, acc, false, false);
            }
        }
        // scaled scores -> LDS (C/D layout: VGPR r -> row r + 8*half, col = lane&15)
        int col = ng * 64 + wave * 16 + arow;
        #pragma unroll
        for (int r = 0; r < 8; ++r)
            Ss[(r + 8 * ahalf) * 256 + col] = acc[r] * 0.03125f; // 1/sqrt(1024)
    }
    __syncthreads();

    // ---------------- Softmax over n<255, result f16 in As (col 255 zeroed)
    {
        int row = tid >> 3, sub = tid & 7;
        float pm = -3.0e38f;
        for (int n = sub; n < NKEY; n += 8) pm = fmaxf(pm, Ss[row * 256 + n]);
        red[row * 8 + sub] = pm;
        __syncthreads();
        float rmax = red[row * 8];
        #pragma unroll
        for (int j = 1; j < 8; ++j) rmax = fmaxf(rmax, red[row * 8 + j]);
        float ps = 0.f;
        for (int n = sub; n < NKEY; n += 8) ps += __expf(Ss[row * 256 + n] - rmax);
        __syncthreads();
        red[row * 8 + sub] = ps;
        __syncthreads();
        float rsum = 0.f;
        #pragma unroll
        for (int j = 0; j < 8; ++j) rsum += red[row * 8 + j];
        float inv = 1.0f / rsum;
        for (int n = sub; n < NKEY; n += 8)
            As[row * AROW + n] = (_Float16)(__expf(Ss[row * 256 + n] - rmax) * inv);
        if (sub == 0) As[row * AROW + 255] = (_Float16)0.0f;
    }

    // ---------------- Pass 2: out = A[16x256] . Vseg[256x1024] (+1 segment roll)
    for (int cpass = 0; cpass < 2; ++cpass) {       // 2 halves of the 1024 columns
        v8f acc[8] = {};
        for (int nch = 0; nch < 8; ++nch) {         // K(=n) chunks of 32
            __syncthreads();
            // stage V chunk TRANSPOSED: Vs[c_local][n_local]; float2 loads,
            // 2x b16 scatter stores (2-way bank conflict at VROW=36)
            #pragma unroll 8
            for (int i = 0; i < 64; ++i) {
                int f = i * 128 + tid;              // float2 index, 0..8191
                int nl = f >> 8;                    // 0..31
                int cp = (f & 255) * 2;             // 0..510
                int seg = nch * 32 + nl + 1;        // seg_v uses segments 1..255
                if (seg > 255) seg = 255;           // clamp pad row (A==0 there)
                int cg = cpass * 512 + cp;
                v2f val = *(const v2f*)&v[base + (size_t)seg * SEG_STRIDE
                                          + (cg >> 6) * L_STRIDE + (cg & 63)];
                Vs[cp * VROW + nl]       = (_Float16)val[0];
                Vs[(cp + 1) * VROW + nl] = (_Float16)val[1];
            }
            __syncthreads();
            V16 a;
            int nb = nch * 32;
            a.h[0] = *(const v8h*)&As[arow * AROW + nb + ahalf * 16];
            a.h[1] = *(const v8h*)&As[arow * AROW + nb + ahalf * 16 + 8];
            #pragma unroll
            for (int ct = 0; ct < 8; ++ct) {        // wave owns 8 col-tiles of 16
                int cl = wave * 128 + ct * 16 + arow;
                V16 bm;
                bm.h4[0] = *(const v4h*)&Vs[cl * VROW + ahalf * 16];
                bm.h4[1] = *(const v4h*)&Vs[cl * VROW + ahalf * 16 + 4];
                bm.h4[2] = *(const v4h*)&Vs[cl * VROW + ahalf * 16 + 8];
                bm.h4[3] = *(const v4h*)&Vs[cl * VROW + ahalf * 16 + 12];
                acc[ct] = __builtin_amdgcn_wmma_f32_16x16x32_f16(
                              false, a.v, false, bm.v, (short)0, acc[ct], false, false);
            }
        }
        // store with roll: tmp segment m -> output segment (m+1) % 256
        #pragma unroll
        for (int ct = 0; ct < 8; ++ct) {
            #pragma unroll
            for (int r = 0; r < 8; ++r) {
                int m = r + 8 * ahalf;
                int cg = cpass * 512 + wave * 128 + ct * 16 + arow;
                int seg_out = (mt * 16 + m + 1) & 255;
                out[(size_t)b * B_STRIDE + (size_t)seg_out * SEG_STRIDE
                    + (cg >> 6) * L_STRIDE + (size_t)h * 64 + (cg & 63)] = acc[ct][r];
            }
        }
    }
}

extern "C" void kernel_launch(void* const* d_in, const int* in_sizes, int n_in,
                              void* d_out, int out_size, void* d_ws, size_t ws_size,
                              hipStream_t stream) {
    const float* q = (const float*)d_in[0];
    const float* k = (const float*)d_in[1];
    const float* v = (const float*)d_in[2];
    // d_in[3] = attn_mask, unused by the reference forward
    float* out = (float*)d_out;
    // B*H*(M/16) = 8*8*16 = 1024 workgroups, 128 threads (4 wave32)
    seg_corr3_kernel<<<dim3(1024), dim3(128), 0, stream>>>(q, k, v, out);
}
